// MoBoAligner_87935160418821
// MI455X (gfx1250) — matrix operand order repaired and checked by reading, not compiled, not run on hardware
//
#include <hip/hip_runtime.h>
#include <hip/hip_bf16.h>

// Problem constants (from setup_inputs)
#define BB 16
#define TT 512
#define MM 2048
#define CC 512
#define NEG_F (-1e30f)

typedef __attribute__((ext_vector_type(16))) __bf16 v16bf;
typedef __attribute__((ext_vector_type(8)))  float  v8f;

union Frag16 { uint4 u[2]; v16bf v; };

// CDNA5 async global->LDS copy (ASYNCcnt-tracked), guarded for portability.
#if defined(__gfx1250__) && __has_builtin(__builtin_amdgcn_global_load_async_to_lds_b128)
#define HAVE_ASYNC_LDS 1
typedef int v4i_t __attribute__((vector_size(16)));
typedef __attribute__((address_space(1))) v4i_t* gv4i_p;
typedef __attribute__((address_space(3))) v4i_t* lv4i_p;
__device__ __forceinline__ void async_cp16(const void* g, void* l) {
    __builtin_amdgcn_global_load_async_to_lds_b128((gv4i_p)g, (lv4i_p)l, 0, 0);
}
__device__ __forceinline__ void async_wait0() {
#if __has_builtin(__builtin_amdgcn_s_wait_asynccnt)
    __builtin_amdgcn_s_wait_asynccnt(0);
#else
    asm volatile("s_wait_asynccnt 0x0" ::: "memory");
#endif
}
#else
#define HAVE_ASYNC_LDS 0
#endif

__device__ __forceinline__ unsigned short f2bf(float f) {
    unsigned u = __float_as_uint(f);
    unsigned r = u + 0x7FFFu + ((u >> 16) & 1u);   // round-to-nearest-even
    return (unsigned short)(r >> 16);
}

__device__ __forceinline__ float lae(float a, float b) {
    // logaddexp, safe for the finite -1e30 sentinel (exp underflows to 0)
    float mx = fmaxf(a, b);
    float mn = fminf(a, b);
    return mx + log1pf(__expf(mn - mx));
}

// ---------------------------------------------------------------- cvt f32->bf16
__global__ void cvt_bf16_kernel(const float* __restrict__ in,
                                unsigned short* __restrict__ out, int n) {
    for (int i = blockIdx.x * blockDim.x + threadIdx.x; i < n;
         i += gridDim.x * blockDim.x)
        out[i] = f2bf(in[i]);
}

// ---------------------------------------------------------------- GEMM1 + gumbel
// energy[b,t,m] = dot(text[b,t,:], mel[b,m,:]) / 512
// log_e = (energy + gumbel(noise)) / temperature
// block: 256 thr (8 waves), tile 128(T) x 128(M), K-chunk 32, LDS stride 48.
// Double-buffered LDS; async global->LDS staging when available.
__global__ __launch_bounds__(256)
void gemm1_loge_kernel(const unsigned short* __restrict__ textb,
                       const unsigned short* __restrict__ melb,
                       const float* __restrict__ noise,
                       const float* __restrict__ tratio,
                       float* __restrict__ log_e) {
    __shared__ unsigned short ldsA[2][128 * 48];
    __shared__ unsigned short ldsB[2][128 * 48];

    const int b  = blockIdx.z;
    const int t0 = blockIdx.y * 128;
    const int m0 = blockIdx.x * 128;
    const int tid  = threadIdx.x;
    const int lane = tid & 31;
    const int wv   = tid >> 5;      // 0..7
    const int hi   = lane >> 4;     // 0/1
    const int ln   = lane & 15;

    const float temp  = 0.1f + 0.9f * tratio[0];
    const float invT  = 1.0f / temp;
    const float scale = 1.0f / 512.0f;    // 1/sqrt(C*Cm)

    v8f zero = {0.f, 0.f, 0.f, 0.f, 0.f, 0.f, 0.f, 0.f};
    v8f acc[8];
#pragma unroll
    for (int n = 0; n < 8; n++) acc[n] = zero;

    const size_t textBase = ((size_t)b * TT + t0) * CC;
    const size_t melBase  = ((size_t)b * MM + m0) * CC;

    // stage one 128x32 bf16 tile for A and B into buffer bi
    auto stage = [&](int kk, int bi) {
        for (int v = tid; v < 512; v += 256) {
            int row = v >> 2, ko = (v & 3) * 8;
            const unsigned short* ga =
                &textb[textBase + (size_t)row * CC + kk + ko];
            const unsigned short* gb =
                &melb[melBase + (size_t)row * CC + kk + ko];
#if HAVE_ASYNC_LDS
            async_cp16(ga, &ldsA[bi][row * 48 + ko]);
            async_cp16(gb, &ldsB[bi][row * 48 + ko]);
#else
            *(uint4*)&ldsA[bi][row * 48 + ko] = *(const uint4*)ga;
            *(uint4*)&ldsB[bi][row * 48 + ko] = *(const uint4*)gb;
#endif
        }
    };

    stage(0, 0);
#if HAVE_ASYNC_LDS
    async_wait0();
#endif
    __syncthreads();

    int cur = 0;
    for (int kk = 0; kk < CC; kk += 32) {
        // prefetch next K-chunk into the other buffer while computing
        if (kk + 32 < CC) stage(kk + 32, cur ^ 1);

        // A fragment (16x32 bf16): lane ln holds row, K = hi*8..+7, 16+hi*8..+7
        Frag16 af;
        const unsigned short* ap = &ldsA[cur][(wv * 16 + ln) * 48 + hi * 8];
        af.u[0] = *(const uint4*)(ap);
        af.u[1] = *(const uint4*)(ap + 16);
#pragma unroll
        for (int n = 0; n < 8; n++) {
            // B fragment (32x16 bf16): lane ln = col, K = hi*8..+7, 16+hi*8..+7
            Frag16 bfr;
            const unsigned short* bp = &ldsB[cur][(n * 16 + ln) * 48 + hi * 8];
            bfr.u[0] = *(const uint4*)(bp);
            bfr.u[1] = *(const uint4*)(bp + 16);
            acc[n] = __builtin_amdgcn_wmma_f32_16x16x32_bf16(
                false, af.v, false, bfr.v, (short)0, acc[n], false, false);
        }

#if HAVE_ASYNC_LDS
        async_wait0();
#endif
        __syncthreads();
        cur ^= 1;
    }

    // epilogue: fused gumbel + temperature
#pragma unroll
    for (int n = 0; n < 8; n++) {
        int mcol = m0 + n * 16 + ln;
#pragma unroll
        for (int r = 0; r < 8; r++) {
            int trow = t0 + wv * 16 + hi * 8 + r;
            size_t idx = ((size_t)b * TT + trow) * MM + mcol;
            float u = noise[idx];
            float g = -__logf(-__logf(u));
            log_e[idx] = (acc[n][r] * scale + g) * invT;
        }
    }
}

// ---------------------------------------------------------------- log_S
// reverse cumulative logsumexp over M per (b,t) row. 256 thr x 8 elems.
__global__ __launch_bounds__(256)
void logS_kernel(const float* __restrict__ log_e, float* __restrict__ log_S) {
    __shared__ float s[2][256];
    const size_t row = blockIdx.x;                 // b*T + t
    const float* x = log_e + row * MM;
    float*       y = log_S + row * MM;
    const int tid = threadIdx.x;
    const int base = tid * 8;

    float v[8], loc[8];
#pragma unroll
    for (int i = 0; i < 8; i++) v[i] = x[base + i];
    loc[7] = v[7];
#pragma unroll
    for (int i = 6; i >= 0; i--) loc[i] = lae(v[i], loc[i + 1]);

    // block-wide inclusive suffix scan of per-thread totals
    int buf = 0;
    s[0][tid] = loc[0];
    __syncthreads();
    for (int off = 1; off < 256; off <<= 1) {
        float val = s[buf][tid];
        if (tid + off < 256) val = lae(val, s[buf][tid + off]);
        s[buf ^ 1][tid] = val;
        __syncthreads();
        buf ^= 1;
    }
    float excl = (tid + 1 < 256) ? s[buf][tid + 1] : NEG_F;
#pragma unroll
    for (int i = 0; i < 8; i++) y[base + i] = lae(loc[i], excl);
}

// ---------------------------------------------------------------- alpha DP
// one block per batch, 1024 threads, 2 columns/thread, sequential over T.
__global__ __launch_bounds__(1024)
void alpha_kernel(const float* __restrict__ log_e,
                  const float* __restrict__ log_S,
                  float* __restrict__ alpha) {
    __shared__ float s[2][1024];
    __shared__ float outSh[MM];
    const int b = blockIdx.x, tid = threadIdx.x;
    const int k0 = tid * 2;
    const size_t baseBM = (size_t)b * TT * MM;

    float a0 = (k0 == 0) ? 0.0f : NEG_F;   // alpha0 = [0, NEG...]
    float a1 = NEG_F;

    for (int t = 0; t < TT; t++) {
        const float* le = log_e + baseBM + (size_t)t * MM;
        const float* ls = log_S + baseBM + (size_t)t * MM;
        float le0 = le[k0], le1 = le[k0 + 1];
        float v0 = a0 - ls[k0], v1 = a1 - ls[k0 + 1];
        float p0 = v0, p1 = lae(v0, v1);

        // forward inclusive prefix scan of thread totals (p1)
        int buf = 0;
        s[0][tid] = p1;
        __syncthreads();
        for (int off = 1; off < 1024; off <<= 1) {
            float val = s[buf][tid];
            if (tid >= off) val = lae(val, s[buf][tid - off]);
            s[buf ^ 1][tid] = val;
            __syncthreads();
            buf ^= 1;
        }
        float excl = (tid > 0) ? s[buf][tid - 1] : NEG_F;
        float o0 = le0 + lae(p0, excl);    // alpha_tail[t][k0]
        float o1 = le1 + lae(p1, excl);

        float* al = alpha + baseBM + (size_t)t * MM;
        al[k0] = o0;  al[k0 + 1] = o1;
        outSh[k0] = o0;  outSh[k0 + 1] = o1;
        __syncthreads();
        // a_new[j] = out[j-1] (j>=1), a_new[0] = NEG
        a0 = (k0 == 0) ? NEG_F : outSh[k0 - 1];
        a1 = outSh[k0];
        __syncthreads();
    }
}

// ---------------------------------------------------------------- beta + gamma
// backward scan over T; fuses gamma = alpha+beta (in place) and LSE over T.
__global__ __launch_bounds__(1024)
void beta_gamma_kernel(const float* __restrict__ log_e,
                       const float* __restrict__ log_S,
                       float* __restrict__ gamma,   // in: alpha, out: gamma
                       float* __restrict__ lseT) {  // [B, M]
    __shared__ float s[2][1024];
    const int b = blockIdx.x, tid = threadIdx.x;
    const int k0 = tid * 2;
    const size_t baseBM = (size_t)b * TT * MM;

    // beta_T: zeros with [M-1] = 1.0 (faithful odd init)
    float bn0 = 0.0f;
    float bn1 = (tid == 1023) ? 1.0f : 0.0f;

    float* gaT = gamma + baseBM + (size_t)(TT - 1) * MM;
    float g0 = gaT[k0] + bn0, g1 = gaT[k0 + 1] + bn1;
    gaT[k0] = g0;  gaT[k0 + 1] = g1;
    float acc0 = g0, acc1 = g1;

    for (int t = TT - 2; t >= 0; t--) {
        const float* le = log_e + baseBM + (size_t)t * MM;
        const float* ls = log_S + baseBM + (size_t)t * MM;
        float w0 = bn0 + le[k0], w1 = bn1 + le[k0 + 1];
        float q1 = w1, q0 = lae(w0, w1);

        // inclusive suffix scan of thread totals (q0)
        __syncthreads();               // guard s reuse across iterations
        int buf = 0;
        s[0][tid] = q0;
        __syncthreads();
        for (int off = 1; off < 1024; off <<= 1) {
            float val = s[buf][tid];
            if (tid + off < 1024) val = lae(val, s[buf][tid + off]);
            s[buf ^ 1][tid] = val;
            __syncthreads();
            buf ^= 1;
        }
        float excl = (tid + 1 < 1024) ? s[buf][tid + 1] : NEG_F;
        float b0 = lae(q0, excl) - ls[k0];
        float b1 = lae(q1, excl) - ls[k0 + 1];

        float* ga = gamma + baseBM + (size_t)t * MM;
        g0 = ga[k0] + b0;  g1 = ga[k0 + 1] + b1;
        ga[k0] = g0;  ga[k0 + 1] = g1;
        acc0 = lae(acc0, g0);  acc1 = lae(acc1, g1);
        bn0 = b0;  bn1 = b1;
    }
    lseT[(size_t)b * MM + k0]     = acc0;
    lseT[(size_t)b * MM + k0 + 1] = acc1;
}

// ---------------------------------------------------------------- gamma_log
__global__ void gammalog_kernel(const float* __restrict__ gamma,
                                const float* __restrict__ lseT,
                                float* __restrict__ out, size_t total) {
    size_t stride = (size_t)gridDim.x * blockDim.x;
    for (size_t i = (size_t)blockIdx.x * blockDim.x + threadIdx.x; i < total;
         i += stride) {
        size_t b = i >> 20;            // / (T*M) = 2^20
        int m = (int)(i & (MM - 1));
        out[i] = gamma[i] - lseT[b * MM + m];
    }
}

// ---------------------------------------------------------------- GEMM2
// expanded[b,m,c] = sum_t gamma[b,t,m] * text[b,t,c]
// A = gamma^T [M x T] (transposed+bf16 on load), B = text^T in LDS [c][t].
// Double-buffered LDS so global loads overlap the WMMA stream.
__global__ __launch_bounds__(256)
void gemm2_kernel(const float* __restrict__ gamma,
                  const unsigned short* __restrict__ textb,
                  float* __restrict__ outExp) {
    __shared__ unsigned short ldsA[2][128 * 48];   // [m_local][t_local]
    __shared__ unsigned short ldsB[2][128 * 48];   // [c_local][t_local]

    const int b  = blockIdx.z;
    const int m0 = blockIdx.y * 128;
    const int c0 = blockIdx.x * 128;
    const int tid  = threadIdx.x;
    const int lane = tid & 31;
    const int wv   = tid >> 5;
    const int hi   = lane >> 4;
    const int ln   = lane & 15;

    v8f zero = {0.f, 0.f, 0.f, 0.f, 0.f, 0.f, 0.f, 0.f};
    v8f acc[8];
#pragma unroll
    for (int n = 0; n < 8; n++) acc[n] = zero;

    // A: gamma[b][kk+tc][m0+mr] -> bf16 ldsA[mr][tc]  (coalesced over mr)
    // B: text_bf16[b][kk+tr][c0+cc] -> ldsB[cc][tr]   (coalesced over cc)
    auto stage = [&](int kk, int bi) {
        for (int i = tid; i < 4096; i += 256) {
            int mr = i & 127, tc = i >> 7;
            float gv = gamma[((size_t)b * TT + kk + tc) * MM + m0 + mr];
            ldsA[bi][mr * 48 + tc] = f2bf(gv);
        }
        for (int i = tid; i < 4096; i += 256) {
            int cc = i & 127, tr = i >> 7;
            ldsB[bi][cc * 48 + tr] =
                textb[((size_t)b * TT + kk + tr) * CC + c0 + cc];
        }
    };

    stage(0, 0);
    __syncthreads();

    int cur = 0;
    for (int kk = 0; kk < TT; kk += 32) {
        if (kk + 32 < TT) stage(kk + 32, cur ^ 1);

        Frag16 af;
        const unsigned short* ap = &ldsA[cur][(wv * 16 + ln) * 48 + hi * 8];
        af.u[0] = *(const uint4*)(ap);
        af.u[1] = *(const uint4*)(ap + 16);
#pragma unroll
        for (int n = 0; n < 8; n++) {
            Frag16 bfr;
            const unsigned short* bp = &ldsB[cur][(n * 16 + ln) * 48 + hi * 8];
            bfr.u[0] = *(const uint4*)(bp);
            bfr.u[1] = *(const uint4*)(bp + 16);
            acc[n] = __builtin_amdgcn_wmma_f32_16x16x32_bf16(
                false, af.v, false, bfr.v, (short)0, acc[n], false, false);
        }
        __syncthreads();
        cur ^= 1;
    }

#pragma unroll
    for (int n = 0; n < 8; n++) {
        int ccol = c0 + n * 16 + ln;
#pragma unroll
        for (int r = 0; r < 8; r++) {
            int mrow = m0 + wv * 16 + hi * 8 + r;
            outExp[((size_t)b * MM + mrow) * CC + ccol] = acc[n][r];
        }
    }
}

// ---------------------------------------------------------------- launch
extern "C" void kernel_launch(void* const* d_in, const int* in_sizes, int n_in,
                              void* d_out, int out_size, void* d_ws,
                              size_t ws_size, hipStream_t stream) {
    const float* text  = (const float*)d_in[0];   // [B,T,C]
    const float* mel   = (const float*)d_in[1];   // [B,M,C]
    const float* noise = (const float*)d_in[2];   // [B,T,M]
    const float* trat  = (const float*)d_in[3];   // [1]
    float* out = (float*)d_out;                   // gamma_log | expanded

    const size_t nText = (size_t)BB * TT * CC;    // 4.19M
    const size_t nMel  = (size_t)BB * MM * CC;    // 16.78M
    const size_t nBTM  = (size_t)BB * TT * MM;    // 16.78M

    // workspace layout (~244 MB)
    char* ws = (char*)d_ws;
    unsigned short* textb = (unsigned short*)ws;
    unsigned short* melb  = (unsigned short*)(ws + nText * 2);
    float* log_e = (float*)(ws + nText * 2 + nMel * 2);
    float* log_S = log_e + nBTM;
    float* gamma = log_S + nBTM;                               // alpha -> gamma
    float* lseT  = gamma + nBTM;                               // [B, M]

    cvt_bf16_kernel<<<2048, 256, 0, stream>>>(text, textb, (int)nText);
    cvt_bf16_kernel<<<4096, 256, 0, stream>>>(mel, melb, (int)nMel);

    gemm1_loge_kernel<<<dim3(MM / 128, TT / 128, BB), 256, 0, stream>>>(
        textb, melb, noise, trat, log_e);

    logS_kernel<<<BB * TT, 256, 0, stream>>>(log_e, log_S);

    alpha_kernel<<<BB, 1024, 0, stream>>>(log_e, log_S, gamma);

    beta_gamma_kernel<<<BB, 1024, 0, stream>>>(log_e, log_S, gamma, lseT);

    gammalog_kernel<<<8192, 256, 0, stream>>>(gamma, lseT, out, nBTM);

    gemm2_kernel<<<dim3(CC / 128, MM / 128, BB), 256, 0, stream>>>(
        gamma, textb, out + nBTM);
}
